// GraphEncoder_18098992185554
// MI455X (gfx1250) — matrix operand order repaired and checked by reading, not compiled, not run on hardware
//
#include <hip/hip_runtime.h>

// ---------------------------------------------------------------------------
// R-GCN encoder for gfx1250 (MI455X): BN -> relational conv (WMMA bf16) ->
// ReLU -> residual, 4 rounds. Matrix math via v_wmma_f32_16x16x32_bf16.
// GEMM: 64 rows x 128 cols per block (8 waves, 4 row-tiles each, B reused),
// software-pipelined double-buffered fragment loads.
// ---------------------------------------------------------------------------

#define F128 128      // feature dim (F_IN == H == 128)
#define RREL 2        // relations
#define KDIM 384      // fused GEMM K = R*H + H
#define KSTEPS 12     // KDIM / 32
#define ROWT 4        // row tiles (of 16) per block
#define BN_EPS 1e-5f
#define BN_BETA 1e-4f

typedef __attribute__((ext_vector_type(16))) __bf16 v16bf;
typedef __attribute__((ext_vector_type(8)))  float  v8f;

static __device__ __forceinline__ unsigned short f2bf(float x) {
    unsigned int u = __float_as_uint(x);
    unsigned int r = (u + 0x7fffu + ((u >> 16) & 1u)) >> 16;  // RNE
    return (unsigned short)r;
}
static __device__ __forceinline__ float bf2f(unsigned short h) {
    return __uint_as_float(((unsigned int)h) << 16);
}

// --------------------------- BatchNorm stats -------------------------------
// stats (floats): [0,128) sum, [128,256) sumsq, [256,384) mu, [384,512) rsig
__global__ void bn_stats_k(const float* __restrict__ in, float* __restrict__ stats,
                           int N) {
    int f = threadIdx.x;                    // 128 threads
    float s = 0.f, sq = 0.f;
    for (int r = blockIdx.x; r < N; r += gridDim.x) {
        float v = in[(size_t)r * F128 + f];
        s += v; sq += v * v;
    }
    atomicAdd(&stats[f], s);
    atomicAdd(&stats[F128 + f], sq);
}

__global__ void bn_finalize_k(float* __restrict__ stats, int N) {
    int f = threadIdx.x;                    // 128 threads, 1 block
    float invN = 1.0f / (float)N;
    float mu  = stats[f] * invN;
    float var = stats[F128 + f] * invN - mu * mu;
    stats[2 * F128 + f] = mu;
    stats[3 * F128 + f] = rsqrtf(var + BN_EPS);
}

// normalize and write bf16 into A-panel columns [256..384)
__global__ void bn_norm_k(const float* __restrict__ in, const float* __restrict__ stats,
                          unsigned short* __restrict__ Apack, int total) {
    int i = blockIdx.x * blockDim.x + threadIdx.x;
    if (i >= total) return;
    int f = i & (F128 - 1);
    int row = i >> 7;
    float v = (in[i] - stats[2 * F128 + f]) * stats[3 * F128 + f] + BN_BETA;
    Apack[(size_t)row * KDIM + 2 * F128 + f] = f2bf(v);
}

// --------------------------- weight panel pack -----------------------------
// Fragment-major B layout: uint index ((t*KSTEPS + kb32)*32 + lane)*8 + r
// where t = column tile (col>>4), lane = (col&15) + 16*hi, hi = (k>>4)&1,
// r = (k&15)>>1, pair slot = k&1. Each lane's 8 regs per K-step contiguous.
__global__ void pack_B_k(const float* __restrict__ W_rel,  // [R,128,128]
                         const float* __restrict__ W_root, // [128,128]
                         unsigned short* __restrict__ Bpack) {
    int col = threadIdx.x;                  // 128
    int k   = blockIdx.x;                   // 384
    float v = (k < RREL * F128) ? W_rel[(size_t)k * F128 + col]
                                : W_root[(size_t)(k - RREL * F128) * F128 + col];
    int t    = col >> 4;
    int lane = (col & 15) + ((k >> 4) & 1) * 16;
    int r    = (k & 15) >> 1;
    int kb32 = k >> 5;
    size_t uidx = (((size_t)t * KSTEPS + kb32) * 32 + lane) * 8 + r;
    Bpack[uidx * 2 + (k & 1)] = f2bf(v);
}

// --------------------------- edge scatter ----------------------------------
// one wave32 per edge; lane handles 4 features via 2 uint (4 bf16) loads.
__global__ void scatter_k(const int* __restrict__ src, const int* __restrict__ dst,
                          const int* __restrict__ et,
                          const unsigned short* __restrict__ Apack,
                          float* __restrict__ sums, float* __restrict__ cnts, int E) {
    int gw   = (blockIdx.x * blockDim.x + threadIdx.x) >> 5;
    int lane = threadIdx.x & 31;
    if (gw >= E) return;
    int s = src[gw], d = dst[gw], t = et[gw];
    const unsigned int* p =
        (const unsigned int*)(Apack + (size_t)s * KDIM + 2 * F128);
    float* acc = sums + ((size_t)d * RREL + t) * F128;
#pragma unroll
    for (int j = 0; j < 2; ++j) {
        unsigned int u = p[lane * 2 + j];
        atomicAdd(&acc[lane * 4 + 2 * j],     bf2f((unsigned short)(u & 0xffffu)));
        atomicAdd(&acc[lane * 4 + 2 * j + 1], bf2f((unsigned short)(u >> 16)));
    }
    if (lane == 0) atomicAdd(&cnts[(size_t)d * RREL + t], 1.0f);
}

// mean = sum / max(cnt,1) -> bf16 into A-panel columns [0..256)
__global__ void mean_pack_k(const float* __restrict__ sums, const float* __restrict__ cnts,
                            unsigned short* __restrict__ Apack, int total) {
    int i = blockIdx.x * blockDim.x + threadIdx.x;
    if (i >= total) return;
    int f  = i & (F128 - 1);
    int nr = i >> 7;                        // n*R + r
    int r  = nr & (RREL - 1);
    int n  = nr >> 1;
    float c = cnts[nr]; c = c > 1.f ? c : 1.f;
    Apack[(size_t)n * KDIM + r * F128 + f] = f2bf(sums[i] / c);
}

// --------------------------- WMMA GEMM + epilogue --------------------------
// C[N x 128] = A[N x 384]_bf16 @ B[384 x 128]_bf16 ; out = relu(C+b) (+resid)
// 256 threads = 8 waves; block covers 64 rows, wave w covers cols [16w,16w+16).
// Each wave holds ROWT accumulators; B fragment reused ROWT times per K-step;
// fragments double-buffered so step k+1 loads overlap step k's WMMAs.
__global__ void gemm_wmma_k(const unsigned short* __restrict__ A,
                            const unsigned short* __restrict__ B,
                            const float* __restrict__ bias,
                            const float* __restrict__ resid,   // may be null
                            float* __restrict__ out, int N) {
    int wave = threadIdx.x >> 5;
    int lane = threadIdx.x & 31;
    int hi   = lane >> 4;                   // K-half select
    int m0   = blockIdx.x * (16 * ROWT);
    int col  = wave * 16 + (lane & 15);

    const unsigned int* Arow[ROWT];
#pragma unroll
    for (int t = 0; t < ROWT; ++t) {
        int row = m0 + t * 16 + (lane & 15);
        if (row >= N) row = N - 1;          // clamp reads; EXEC stays all-ones
        Arow[t] = (const unsigned int*)(A + (size_t)row * KDIM);
    }
    // B fragment base for this wave/lane: 8 contiguous uints per K-step.
    const unsigned int* Bp =
        (const unsigned int*)B + ((size_t)wave * KSTEPS * 32 + lane) * 8;

    union Frag { unsigned int u[8]; v16bf v; };
    Frag bfr[2];
    Frag afr[2][ROWT];

    // A 16x32 bf16 fragment: lanes 0-15 K={0..15},
    // lanes 16-31 K={16..31}; each 32-bit reg holds one K-pair.
#define LOAD_B(buf, step)                                                    \
    _Pragma("unroll") for (int r = 0; r < 8; ++r)                            \
        bfr[buf].u[r] = Bp[(size_t)(step) * 256 + r];
#define LOAD_A(buf, step)                                                    \
    _Pragma("unroll") for (int t = 0; t < ROWT; ++t)                         \
        _Pragma("unroll") for (int r = 0; r < 8; ++r) {                      \
            int ka = (step) * 32 + 2 * r                                     \
                     + hi * 16;                                              \
            afr[buf][t].u[r] = Arow[t][ka >> 1];                             \
        }

    v8f acc[ROWT] = {};
    LOAD_B(0, 0)
    LOAD_A(0, 0)
#pragma unroll
    for (int kb32 = 0; kb32 < KSTEPS; ++kb32) {
        int cur = kb32 & 1;
        int nxt = cur ^ 1;
        if (kb32 + 1 < KSTEPS) {            // prefetch next step's fragments
            LOAD_B(nxt, kb32 + 1)
            LOAD_A(nxt, kb32 + 1)
        }
#pragma unroll
        for (int t = 0; t < ROWT; ++t)
            acc[t] = __builtin_amdgcn_wmma_f32_16x16x32_bf16(
                false, afr[cur][t].v, false, bfr[cur].v, (short)0, acc[t],
                false, false);
    }
#undef LOAD_A
#undef LOAD_B

    float bv = bias[col];
#pragma unroll
    for (int t = 0; t < ROWT; ++t) {
#pragma unroll
        for (int r = 0; r < 8; ++r) {
            int orow = m0 + t * 16 + r + hi * 8;  // C layout: hi lanes -> M=8+r
            if (orow >= N) continue;
            float v = acc[t][r] + bv;
            v = v > 0.f ? v : 0.f;
            size_t idx = (size_t)orow * F128 + col;
            if (resid) v += resid[idx];
            out[idx] = v;
        }
    }
}

// ---------------------------------------------------------------------------

static inline size_t align256(size_t x) { return (x + 255u) & ~(size_t)255u; }

extern "C" void kernel_launch(void* const* d_in, const int* in_sizes, int n_in,
                              void* d_out, int out_size, void* d_ws, size_t ws_size,
                              hipStream_t stream) {
    const int N = in_sizes[0] / F128;
    const int E = in_sizes[2];
    const int L = 3;

    const float* x       = (const float*)d_in[0];
    const int*   eidx    = (const int*)d_in[1];
    const int*   etype   = (const int*)d_in[2];
    const float* W0_rel  = (const float*)d_in[3];
    const float* W0_root = (const float*)d_in[4];
    const float* b0      = (const float*)d_in[5];
    const float* Ws_rel  = (const float*)d_in[6];
    const float* Ws_root = (const float*)d_in[7];
    const float* bs      = (const float*)d_in[8];

    const int* src = eidx;
    const int* dst = eidx + E;

    // workspace layout: [sums][cnts][stats] (zeroed together) [h][Apack][Bpack]
    char* ws = (char*)d_ws;
    size_t off = 0;
    float* sums = (float*)(ws + off);  off = align256(off + (size_t)N * RREL * F128 * 4);
    float* cnts = (float*)(ws + off);  off = align256(off + (size_t)N * RREL * 4);
    float* stats = (float*)(ws + off); off = align256(off + 4 * F128 * 4);
    size_t zero_bytes = off;
    float* h = (float*)(ws + off);     off = align256(off + (size_t)N * F128 * 4);
    unsigned short* Apack = (unsigned short*)(ws + off);
    off = align256(off + (size_t)N * KDIM * 2);
    unsigned short* Bpack = (unsigned short*)(ws + off);
    off = align256(off + (size_t)KDIM * F128 * 2);
    (void)ws_size; (void)n_in; (void)out_size;

    const int normTotal = N * F128;
    const int meanTotal = N * RREL * F128;
    const int scatterBlocks = (E * 32 + 255) / 256;
    const int gemmBlocks = (N + 16 * ROWT - 1) / (16 * ROWT);

    for (int layer = 0; layer <= L; ++layer) {
        const float* in     = (layer == 0) ? x : h;
        const float* W_rel  = (layer == 0) ? W0_rel  : Ws_rel  + (size_t)(layer - 1) * RREL * F128 * F128;
        const float* W_root = (layer == 0) ? W0_root : Ws_root + (size_t)(layer - 1) * F128 * F128;
        const float* bias   = (layer == 0) ? b0      : bs      + (size_t)(layer - 1) * F128;
        const float* resid  = (layer == 0) ? nullptr : h;
        float* out = (layer == L) ? (float*)d_out : h;

        hipMemsetAsync(sums, 0, zero_bytes, stream);

        bn_stats_k<<<256, F128, 0, stream>>>(in, stats, N);
        bn_finalize_k<<<1, F128, 0, stream>>>(stats, N);
        bn_norm_k<<<(normTotal + 255) / 256, 256, 0, stream>>>(in, stats, Apack, normTotal);
        pack_B_k<<<KDIM, F128, 0, stream>>>(W_rel, W_root, Bpack);
        scatter_k<<<scatterBlocks, 256, 0, stream>>>(src, dst, etype, Apack, sums, cnts, E);
        mean_pack_k<<<(meanTotal + 255) / 256, 256, 0, stream>>>(sums, cnts, Apack, meanTotal);
        gemm_wmma_k<<<gemmBlocks, 256, 0, stream>>>(Apack, Bpack, bias, resid, out, N);
    }
}
